// OptimizedMoEBlock_57561151701418
// MI455X (gfx1250) — matrix operand order repaired and checked by reading, not compiled, not run on hardware
//
#include <hip/hip_runtime.h>

typedef __attribute__((ext_vector_type(16))) _Float16 v16h;
typedef __attribute__((ext_vector_type(8)))  _Float16 v8h;
typedef __attribute__((ext_vector_type(4)))  _Float16 v4h;
typedef __attribute__((ext_vector_type(8)))  float    v8f;

#define N_TOK 8192
#define DIM   512
#define FF    2048
#define NEXP  8
#define TOPK  2
#define CAP   (N_TOK * TOPK)     // worst-case entries per expert
#define TILE_M 64
#define NTILES (CAP / TILE_M)    // 256 tiles per expert

union AF { v16h v; v8h h[2]; };

// ---------------- fp32 -> fp16 conversion ----------------
__global__ void cvt_f32_f16(const float* __restrict__ s, _Float16* __restrict__ d, int n) {
    int i = (blockIdx.x * blockDim.x + threadIdx.x) * 4;
    if (i >= n) return;
    float4 v = *(const float4*)(s + i);
    v4h o = { (_Float16)v.x, (_Float16)v.y, (_Float16)v.z, (_Float16)v.w };
    *(v4h*)(d + i) = o;
}

// ---------------- routing ----------------
__global__ void route_zero(int* counts) {
    if (threadIdx.x < NEXP) counts[threadIdx.x] = 0;
}

__global__ void route_build(const int* __restrict__ eidx, int* counts, int* lists) {
    int i = blockIdx.x * blockDim.x + threadIdx.x;
    if (i >= N_TOK * TOPK) return;
    int e = eidx[i];
    int pos = atomicAdd(&counts[e], 1);
    lists[e * CAP + pos] = i;            // slot id = n*TOPK + k
}

// ---------------- grouped MoE FFN (WMMA f16) ----------------
// Block = 8 waves = 256 threads, handles 64 token-expert pairs of one expert.
__global__ __launch_bounds__(256)
void moe_ffn(const _Float16* __restrict__ xh,
             const _Float16* __restrict__ w_up_h,
             const float* __restrict__ b_up,
             const _Float16* __restrict__ w_down_h,
             const float* __restrict__ b_down,
             const int* __restrict__ counts,
             const int* __restrict__ lists,
             float* __restrict__ y)
{
    __shared__ _Float16 Hs[TILE_M][FF + 8];    // silu(up) activations, ~263 KB (CDNA5 LDS)
    __shared__ int slotIds[TILE_M];

    const int e    = blockIdx.x >> 8;          // / NTILES
    const int tile = blockIdx.x & (NTILES - 1);
    const int cnt  = counts[e];
    const int tstart = tile * TILE_M;
    if (tstart >= cnt) return;

    if (threadIdx.x < TILE_M) {
        int ent = tstart + threadIdx.x;
        slotIds[threadIdx.x] = (ent < cnt) ? lists[e * CAP + ent] : -1;
    }
    __syncthreads();

    const int wave  = threadIdx.x >> 5;        // 0..7
    const int lane  = threadIdx.x & 31;
    const int ln    = lane & 15;
    const int hi    = (lane >= 16) ? 1 : 0;
    const int aoff  = hi ? 16 : 0;             // A-frag K sub-offset (contiguous K, matches B)
    const int boff  = hi ? 16 : 0;             // B-frag K sub-offset (contiguous K, matches A)
    const int mb    = hi ? 8 : 0;              // C-frag row base

    // per-lane A row pointers: row (mt*16 + ln) of the tile -> token row of xh
    const _Float16* aptr[4];
    #pragma unroll
    for (int mt = 0; mt < 4; ++mt) {
        int sid = slotIds[mt * 16 + ln];
        int tok = (sid >= 0) ? (sid >> 1) : 0;
        aptr[mt] = xh + (size_t)tok * DIM;
    }

    // ================= phase 1: H = silu(X * Wup^T + b_up) =================
    {
        const _Float16* wu = w_up_h + (size_t)e * FF * DIM;
        for (int t = 0; t < 8; ++t) {          // each wave owns 256 f-cols: 8 iters x 2 tiles
            const int n0    = wave * 256 + t * 32;
            const int fcol0 = n0 + ln;
            const int fcol1 = n0 + 16 + ln;
            const _Float16* brow0 = wu + (size_t)fcol0 * DIM;
            const _Float16* brow1 = wu + (size_t)fcol1 * DIM;
            v8f c0[4] = {}, c1[4] = {};
            for (int k0 = 0; k0 < DIM; k0 += 32) {
                AF b0, b1;
                b0.h[0] = *(const v8h*)&brow0[k0 + boff];
                b0.h[1] = *(const v8h*)&brow0[k0 + boff + 8];
                b1.h[0] = *(const v8h*)&brow1[k0 + boff];
                b1.h[1] = *(const v8h*)&brow1[k0 + boff + 8];
                #pragma unroll
                for (int mt = 0; mt < 4; ++mt) {
                    AF a;
                    a.h[0] = *(const v8h*)&aptr[mt][k0 + aoff];
                    a.h[1] = *(const v8h*)&aptr[mt][k0 + aoff + 8];
                    c0[mt] = __builtin_amdgcn_wmma_f32_16x16x32_f16(false, a.v, false, b0.v,
                                                                    (short)0, c0[mt], false, false);
                    c1[mt] = __builtin_amdgcn_wmma_f32_16x16x32_f16(false, a.v, false, b1.v,
                                                                    (short)0, c1[mt], false, false);
                }
            }
            const float bias0 = b_up[e * FF + fcol0];
            const float bias1 = b_up[e * FF + fcol1];
            #pragma unroll
            for (int mt = 0; mt < 4; ++mt) {
                #pragma unroll
                for (int r = 0; r < 8; ++r) {
                    float h0 = c0[mt][r] + bias0;
                    float h1 = c1[mt][r] + bias1;
                    h0 = h0 / (1.0f + __expf(-h0));     // silu
                    h1 = h1 / (1.0f + __expf(-h1));
                    Hs[mt * 16 + mb + r][fcol0] = (_Float16)h0;
                    Hs[mt * 16 + mb + r][fcol1] = (_Float16)h1;
                }
            }
        }
    }
    __syncthreads();

    // ================= phase 2: Y = H * Wdown^T + b_down, scatter =================
    {
        const _Float16* wd = w_down_h + (size_t)e * DIM * FF;
        const _Float16* brow[4];
        #pragma unroll
        for (int t = 0; t < 4; ++t)            // each wave owns 64 d-cols: 4 tiles
            brow[t] = wd + (size_t)(wave * 64 + t * 16 + ln) * FF;

        v8f cd[4][4] = {};                     // [d-tile][m-tile]
        for (int k0 = 0; k0 < FF; k0 += 32) {
            AF a[4], b[4];
            #pragma unroll
            for (int mt = 0; mt < 4; ++mt) {
                a[mt].h[0] = *(const v8h*)&Hs[mt * 16 + ln][k0 + aoff];
                a[mt].h[1] = *(const v8h*)&Hs[mt * 16 + ln][k0 + aoff + 8];
            }
            #pragma unroll
            for (int t = 0; t < 4; ++t) {
                b[t].h[0] = *(const v8h*)&brow[t][k0 + boff];
                b[t].h[1] = *(const v8h*)&brow[t][k0 + boff + 8];
            }
            #pragma unroll
            for (int t = 0; t < 4; ++t)
                #pragma unroll
                for (int mt = 0; mt < 4; ++mt)
                    cd[t][mt] = __builtin_amdgcn_wmma_f32_16x16x32_f16(false, a[mt].v, false, b[t].v,
                                                                       (short)0, cd[t][mt], false, false);
        }

        #pragma unroll
        for (int t = 0; t < 4; ++t) {
            const int dcol = wave * 64 + t * 16 + ln;
            const float bias = b_down[e * DIM + dcol];
            #pragma unroll
            for (int mt = 0; mt < 4; ++mt) {
                #pragma unroll
                for (int r = 0; r < 8; ++r) {
                    int s = slotIds[mt * 16 + mb + r];
                    if (s >= 0) y[(size_t)s * DIM + dcol] = cd[t][mt][r] + bias;
                }
            }
        }
    }
}

// ---------------- weighted combine ----------------
__global__ void combine_k(const float* __restrict__ y, const float* __restrict__ rw,
                          float* __restrict__ out) {
    int i = blockIdx.x * blockDim.x + threadIdx.x;   // over N*DIM/4
    if (i >= N_TOK * DIM / 4) return;
    int n  = i / (DIM / 4);
    int dq = i % (DIM / 4);
    float w0 = rw[n * 2 + 0];
    float w1 = rw[n * 2 + 1];
    float4 a = *((const float4*)(y + (size_t)(n * 2 + 0) * DIM) + dq);
    float4 b = *((const float4*)(y + (size_t)(n * 2 + 1) * DIM) + dq);
    float4 o = { w0 * a.x + w1 * b.x, w0 * a.y + w1 * b.y,
                 w0 * a.z + w1 * b.z, w0 * a.w + w1 * b.w };
    ((float4*)out)[i] = o;
}

extern "C" void kernel_launch(void* const* d_in, const int* in_sizes, int n_in,
                              void* d_out, int out_size, void* d_ws, size_t ws_size,
                              hipStream_t stream) {
    const float* x      = (const float*)d_in[0];
    const int*   eidx   = (const int*)  d_in[1];
    const float* rw     = (const float*)d_in[2];
    const float* w_up   = (const float*)d_in[3];
    const float* b_up   = (const float*)d_in[4];
    const float* w_down = (const float*)d_in[5];
    const float* b_down = (const float*)d_in[6];
    float* out = (float*)d_out;

    char* ws = (char*)d_ws;
    _Float16* w_up_h   = (_Float16*)(ws);                        // 16 MiB
    _Float16* w_down_h = (_Float16*)(ws + ((size_t)16 << 20));   // 16 MiB
    _Float16* xh       = (_Float16*)(ws + ((size_t)32 << 20));   //  8 MiB
    float*    y        = (float*)   (ws + ((size_t)40 << 20));   // 32 MiB
    int*      counts   = (int*)     (ws + ((size_t)72 << 20));   // 32 B
    int*      lists    = (int*)     (ws + ((size_t)72 << 20) + 256); // 512 KiB

    const int nw = NEXP * FF * DIM;                              // 8.4M elems each
    const int nx = N_TOK * DIM;                                  // 4.2M elems
    cvt_f32_f16<<<(nw / 4 + 255) / 256, 256, 0, stream>>>(w_up,   w_up_h,   nw);
    cvt_f32_f16<<<(nw / 4 + 255) / 256, 256, 0, stream>>>(w_down, w_down_h, nw);
    cvt_f32_f16<<<(nx / 4 + 255) / 256, 256, 0, stream>>>(x,      xh,       nx);
    route_zero <<<1, 64, 0, stream>>>(counts);
    route_build<<<(N_TOK * TOPK + 255) / 256, 256, 0, stream>>>(eidx, counts, lists);
    moe_ffn    <<<NEXP * NTILES, 256, 0, stream>>>(xh, w_up_h, b_up, w_down_h, b_down,
                                                   counts, lists, y);
    combine_k  <<<(N_TOK * DIM / 4 + 255) / 256, 256, 0, stream>>>(y, rw, out);
}
